// SparseConv_85177791414438
// MI455X (gfx1250) — matrix operand order repaired and checked by reading, not compiled, not run on hardware
//
#include <hip/hip_runtime.h>
#include <hip/hip_bf16.h>

// ---------------- problem constants (match reference) ----------------
#define HH 240
#define WW 304
#define IN_CH 16
#define CINP  32          // CIN=18 padded to 32 for WMMA K-tiling
#define OUT_CH 32
#define HW (HH * WW)      // 72960, multiple of 16

typedef __attribute__((ext_vector_type(16))) _Float16 v16h;
typedef __attribute__((ext_vector_type(8)))  _Float16 v8h;
typedef __attribute__((ext_vector_type(8)))  float    v8f;
typedef __attribute__((ext_vector_type(4)))  unsigned int v4u;
typedef __attribute__((ext_vector_type(8)))  int      v8i;
typedef __attribute__((ext_vector_type(4)))  int      v4i;

#define WBYTES (9 * 2 * 32 * 16 * 2)   // 18432 bytes of swizzled weight fragments

// ---------------- helpers ----------------
__device__ __forceinline__ int event_key(const float4 ev, int i,
                                         const int* __restrict__ offs, int nb) {
    int batch = 0;
    for (int j = 0; j < nb; ++j) batch += (i >= offs[j]) ? 1 : 0;   // searchsorted right
    int yi = (int)rintf(ev.y * (float)HH);
    int xi = (int)rintf(ev.x * (float)WW);
    yi = min(max(yi, 0), HH - 1);
    xi = min(max(xi, 0), WW - 1);
    return (batch * HH + yi) * WW + xi;
}

// ---------------- kernel 1: zero the fp32 accumulator + zero page ----------------
__global__ void k_zero(float4* __restrict__ acc4, int n4, float4* __restrict__ zp4) {
    int t = blockIdx.x * blockDim.x + threadIdx.x;
    float4 z = {0.f, 0.f, 0.f, 0.f};
    if (t < n4) acc4[t] = z;
    if (t < 8)  zp4[t] = z;      // 128-byte zero page (f16 zeros too)
}

// ---------------- kernel 2: swizzle weights into per-lane B fragments ----------------
// wf16 flat index = ((tap*2 + nt)*32 + lane)*16 + e ;  K = (lane>>4)*8 + (e&7) + (e>>3)*16, oc = nt*16 + (lane&15)
__global__ void k_prep_weights(const float* __restrict__ w, _Float16* __restrict__ wf16) {
    for (int i = threadIdx.x; i < 9 * 2 * 32 * 16; i += blockDim.x) {
        int tap  = i >> 10;            // /1024
        int nt   = (i >> 9) & 1;
        int lane = (i >> 4) & 31;
        int e    = i & 15;
        int kk   = ((lane >> 4) << 3) + (e & 7) + ((e >> 3) << 4);          // channel index 0..31
        int oc   = (nt << 4) + (lane & 15);
        float v  = (kk < 18) ? w[(tap * 18 + kk) * OUT_CH + oc] : 0.f;
        wf16[i]  = (_Float16)v;
    }
}

// ---------------- kernel 3: scatter (segment-sum via fp32 atomics) ----------------
__global__ void k_scatter(const float4* __restrict__ events4,
                          const float4* __restrict__ feat4,
                          float* __restrict__ acc,          // [npix][20]: 18ch + count + pad
                          const int* __restrict__ offs, int nb, int n) {
    int i = blockIdx.x * blockDim.x + threadIdx.x;
    if (i >= n) return;
    float4 ev = events4[i];
    int key = event_key(ev, i, offs, nb);
    float* row = acc + (long)key * 20;
    float pos = ev.w;
    atomicAdd(row + 0, pos);
    atomicAdd(row + 1, 1.f - pos);
#pragma unroll
    for (int q = 0; q < 4; ++q) {
        float4 f = feat4[(long)i * 4 + q];
        atomicAdd(row + 2 + q * 4 + 0, f.x);
        atomicAdd(row + 2 + q * 4 + 1, f.y);
        atomicAdd(row + 2 + q * 4 + 2, f.z);
        atomicAdd(row + 2 + q * 4 + 3, f.w);
    }
    atomicAdd(row + 18, 1.f);
}

// ---------------- kernel 4: normalize + convert to f16, pad 18 -> 32 channels ----------------
__global__ void k_normalize(const float* __restrict__ acc,
                            _Float16* __restrict__ dense, int npix) {
    int p = blockIdx.x * blockDim.x + threadIdx.x;
    if (p >= npix) return;
    const float* row = acc + (long)p * 20;
    float s = 1.f / fmaxf(row[18], 1.f);
    v16h t0, t1;
#pragma unroll
    for (int c = 0; c < 16; ++c) t0[c] = (_Float16)(row[c] * s);
    t1[0] = (_Float16)(row[16] * s);
    t1[1] = (_Float16)(row[17] * s);
#pragma unroll
    for (int c = 2; c < 16; ++c) t1[c] = (_Float16)0.f;
    v16h* dst = (v16h*)(dense + (long)p * CINP);
    dst[0] = t0;
    dst[1] = t1;
}

// ---------------- kernel 5: implicit-GEMM 3x3 conv via WMMA ----------------
// One wave computes 16 pixels x 32 output channels.  K = 9 taps x 32 ch = 9 WMMA per N-tile.
// Weight fragments are staged into LDS by the Tensor Data Mover (one DMA per block,
// tracked by TENSORcnt); the real LDS offset of ldsW goes into the descriptor so the
// compiler sees the array as captured/written.
__global__ void k_conv_wmma(const _Float16* __restrict__ dense,
                            const _Float16* __restrict__ wf16,
                            const _Float16* __restrict__ zpage,
                            float* __restrict__ out32, int npix) {
    __shared__ __align__(16) _Float16 ldsW[9 * 2 * 32 * 16];   // 18 KB of B fragments

#if defined(__gfx1250__) && __has_builtin(__builtin_amdgcn_tensor_load_to_lds)
    {
        // LDS byte offset of ldsW: flat LDS addresses carry the offset in addr[31:0]
        const unsigned lds_off = (unsigned)(size_t)(const void*)ldsW;
        if (threadIdx.x < 32) {             // wave 0 issues one TDM DMA for all 18432 B
            // D# group0: count=1 | lds_addr (bits 63:32) | global_addr (bits 120:64) | type=2
            unsigned long long ga = (unsigned long long)(uintptr_t)wf16;
            v4u g0 = { 1u,                                     // count=1, gather off
                       lds_off,                                // lds_addr
                       (unsigned int)(ga & 0xFFFFFFFFull),
                       (unsigned int)((ga >> 32) & 0x01FFFFFFull) | (2u << 30) };
            // D# group1: 1 row of 2304 x 8-byte elements (18432 B), stride 2304
            const unsigned int NE = WBYTES / 8;                // 2304
            v8i g1 = { (int)(3u << 16),                        // data_size = 8B
                       (int)((NE & 0xFFFFu) << 16),            // tensor_dim0 lo
                       (int)(((NE >> 16) & 0xFFFFu) | (1u << 16)),  // dim0 hi | tensor_dim1=1
                       (int)((NE & 0xFFFFu) << 16),            // tile_dim0 = 2304
                       (int)1,                                 // tile_dim1 = 1, tile_dim2 = 0
                       (int)NE,                                // tensor_dim0_stride lo
                       0, 0 };
            v4i g2 = { 0, 0, 0, 0 };
            v4i g3 = { 0, 0, 0, 0 };
#if __clang_major__ >= 23
            v8i g4 = { 0, 0, 0, 0, 0, 0, 0, 0 };
            __builtin_amdgcn_tensor_load_to_lds(g0, g1, g2, g3, g4, 0);
#else
            __builtin_amdgcn_tensor_load_to_lds(g0, g1, g2, g3, 0);
#endif
            __builtin_amdgcn_s_wait_tensorcnt(0);
        }
        // compiler barrier: the DMA above wrote ldsW (captured via lds_off)
        asm volatile("" ::: "memory");
    }
#else
    {   // fallback: cooperative stage of all B fragments into LDS
        const uint4* src = (const uint4*)wf16;
        uint4* dst = (uint4*)ldsW;
        for (int i = threadIdx.x; i < WBYTES / 16; i += blockDim.x) dst[i] = src[i];
    }
#endif
    __syncthreads();

    const int lane  = threadIdx.x & 31;
    const int wave  = threadIdx.x >> 5;
    const int tile  = blockIdx.x * (blockDim.x >> 5) + wave;
    const int pb    = tile << 4;                 // first pixel of tile
    if (pb >= npix) return;                      // uniform per wave -> EXEC all ones

    const int m     = lane & 15;
    const int khalf = lane >> 4;
    const int p     = pb + m;
    const int b     = p / HW;
    const int rem   = p - b * HW;
    const int y     = rem / WW;
    const int x     = rem - y * WW;

    v8f c0 = {}, c1 = {};
#pragma unroll
    for (int tap = 0; tap < 9; ++tap) {
        const int dy = tap / 3 - 1, dx = tap % 3 - 1;
        const int ny = y + dy, nx = x + dx;
        const bool ok = ((unsigned)ny < (unsigned)HH) && ((unsigned)nx < (unsigned)WW);
        const _Float16* ap = ok ? (dense + ((long)((b * HH + ny) * WW + nx) << 5)) : zpage;
        // A fragment (16-bit A 16x32 layout): e=0..7 -> K=khalf*8+e ; e=8..15 -> K=16+khalf*8+(e-8)
        v8h alo = *(const v8h*)(ap + khalf * 8);
        v8h ahi = *(const v8h*)(ap + 16 + khalf * 8);
        v16h a;
#pragma unroll
        for (int i = 0; i < 8; ++i) { a[i] = alo[i]; a[i + 8] = ahi[i]; }

        v16h b0 = *(const v16h*)(&ldsW[(tap * 2 + 0) * 512 + lane * 16]);
        v16h b1 = *(const v16h*)(&ldsW[(tap * 2 + 1) * 512 + lane * 16]);

        c0 = __builtin_amdgcn_wmma_f32_16x16x32_f16(false, a, false, b0, (short)0, c0, false, false);
        c1 = __builtin_amdgcn_wmma_f32_16x16x32_f16(false, a, false, b1, (short)0, c1, false, false);
    }

    // C/D layout: lane holds column oc = nt*16 + (lane&15); VGPR r holds row M = khalf*8 + r
    const int ocb  = lane & 15;
    const long row0 = pb + khalf * 8;
#pragma unroll
    for (int r = 0; r < 8; ++r)
        out32[(row0 + r) * OUT_CH + ocb] = c0[r];
#pragma unroll
    for (int r = 0; r < 8; ++r)
        out32[(row0 + r) * OUT_CH + 16 + ocb] = c1[r];
}

// ---------------- kernel 6: gather out[key] + bias ----------------
__global__ void k_gather(const float4* __restrict__ events4,
                         const float4* __restrict__ out4,     // conv result, 8 float4 / pixel
                         const float4* __restrict__ bias4,
                         float4* __restrict__ dst4,
                         const int* __restrict__ offs, int nb, int n) {
    int t = blockIdx.x * blockDim.x + threadIdx.x;
    int e = t >> 2;
    if (e >= n) return;
    int sub = t & 3;
    float4 ev = events4[e];
    int key = event_key(ev, e, offs, nb);
    float4 o0 = out4[(long)key * 8 + sub * 2 + 0];
    float4 o1 = out4[(long)key * 8 + sub * 2 + 1];
    float4 b0 = bias4[sub * 2 + 0];
    float4 b1 = bias4[sub * 2 + 1];
    o0.x += b0.x; o0.y += b0.y; o0.z += b0.z; o0.w += b0.w;
    o1.x += b1.x; o1.y += b1.y; o1.z += b1.z; o1.w += b1.w;
    dst4[(long)e * 8 + sub * 2 + 0] = o0;
    dst4[(long)e * 8 + sub * 2 + 1] = o1;
}

// ---------------- host launcher ----------------
extern "C" void kernel_launch(void* const* d_in, const int* in_sizes, int n_in,
                              void* d_out, int out_size, void* d_ws, size_t ws_size,
                              hipStream_t stream) {
    const float* events   = (const float*)d_in[0];
    const float* features = (const float*)d_in[1];
    const float* weight   = (const float*)d_in[2];
    const float* bias     = (const float*)d_in[3];
    const int*   offsets  = (const int*)d_in[4];
    const int nb  = in_sizes[4];
    const int n   = in_sizes[0] / 4;
    const int npix = nb * HW;

    // workspace layout (bytes):
    //   [0, npix*32*4)                 : out32 (f32 conv output)  -- also aliases acc (dead after normalize)
    //   [offD, +npix*32*2)             : dense f16, 32ch padded
    //   [offW, +18432)                 : swizzled f16 weight fragments
    //   [offZ, +128)                   : zero page
    char* ws = (char*)d_ws;
    const size_t szOut = (size_t)npix * OUT_CH * sizeof(float);
    const size_t offD  = (szOut + 255) & ~(size_t)255;
    const size_t offW  = (offD + (size_t)npix * CINP * sizeof(_Float16) + 255) & ~(size_t)255;
    const size_t offZ  = (offW + WBYTES + 255) & ~(size_t)255;

    float*     acc    = (float*)ws;                       // [npix][20]
    float*     out32  = (float*)ws;
    _Float16*  dense  = (_Float16*)(ws + offD);
    _Float16*  wf16   = (_Float16*)(ws + offW);
    _Float16*  zpage  = (_Float16*)(ws + offZ);

    const int n4acc = npix * 20 / 4;
    k_zero<<<(n4acc + 255) / 256, 256, 0, stream>>>((float4*)acc, n4acc, (float4*)zpage);
    k_prep_weights<<<1, 256, 0, stream>>>(weight, wf16);
    k_scatter<<<(n + 255) / 256, 256, 0, stream>>>((const float4*)events, (const float4*)features,
                                                   acc, offsets, nb, n);
    k_normalize<<<(npix + 255) / 256, 256, 0, stream>>>(acc, dense, npix);

    const int ntiles = (npix + 15) / 16;                  // one wave per 16-pixel tile
    const int wavesPerBlk = 8;                            // 256 threads, wave32
    k_conv_wmma<<<(ntiles + wavesPerBlk - 1) / wavesPerBlk, 256, 0, stream>>>(
        dense, wf16, zpage, out32, npix);

    k_gather<<<(4 * n + 255) / 256, 256, 0, stream>>>((const float4*)events, (const float4*)out32,
                                                      (const float4*)bias, (float4*)d_out,
                                                      offsets, nb, n);
}
